// SelfAttention_82952998355599
// MI455X (gfx1250) — compile-verified, matrix operand order checked by reading, not run on hardware
//
#include <hip/hip_runtime.h>
#include <hip/hip_bf16.h>
#include <cstdint>

typedef _Float16 f16;
typedef __attribute__((ext_vector_type(16))) _Float16 v16h;
typedef __attribute__((ext_vector_type(8)))  float    v8f;

#define BB 4
#define SS 2048
#define EE 1024
#define HH 16
#define DD 64

// ---------------------------------------------------------------------------
// Operand gathers per CDNA5 WMMA VGPR layouts (ISA 7.12.2, wave32):
//  A (16x32 f16): lane L holds row m=L&15; half-pair r holds
//      K = (r<4 ? 2r : 16+2(r-4)) + 8*(L>>4), K+1
//  B (32x16 f16): same gather over the row-major TRANSPOSED B source
//      (row index = output column n = L&15).
// ---------------------------------------------------------------------------
static __device__ __forceinline__ v16h load_tile_f16(const f16* base, int ld) {
    const int lane = threadIdx.x & 31;
    const int m = lane & 15, hi = lane >> 4;
    const f16* row = base + (size_t)m * ld;
    v16h out;
#pragma unroll
    for (int r = 0; r < 8; ++r) {
        int k = ((r < 4) ? (2 * r) : (16 + 2 * (r - 4))) + 8 * hi;
        out[2 * r]     = row[k];
        out[2 * r + 1] = row[k + 1];
    }
    return out;
}

static __device__ __forceinline__ v16h load_tile_f32(const float* base, int ld) {
    const int lane = threadIdx.x & 31;
    const int m = lane & 15, hi = lane >> 4;
    const float* row = base + (size_t)m * ld;
    v16h out;
#pragma unroll
    for (int r = 0; r < 8; ++r) {
        int k = ((r < 4) ? (2 * r) : (16 + 2 * (r - 4))) + 8 * hi;
        out[2 * r]     = (f16)row[k];
        out[2 * r + 1] = (f16)row[k + 1];
    }
    return out;
}

static __device__ __forceinline__ v8f wmma_f16(v16h a, v16h b, v8f c) {
    return __builtin_amdgcn_wmma_f32_16x16x32_f16(false, a, false, b,
                                                  (short)0, c, false, false);
}

// Async global->LDS copy of 16 bytes, tracked by ASYNCcnt (CDNA5 path;
// ISA 15.18.3 opcode 98). LDS dest offset comes from the low 32 bits of
// the generic pointer (LDS aperture: addr[31:0] is the LDS byte address).
static __device__ __forceinline__ void async_b128(const f16* gsrc, f16* ldst) {
    unsigned           dst = (unsigned)(uintptr_t)(void*)ldst;
    unsigned long long src = (unsigned long long)(uintptr_t)gsrc;
    asm volatile("global_load_async_to_lds_b128 %0, %1, off"
                 :: "v"(dst), "v"(src) : "memory");
}
static __device__ __forceinline__ void wait_async0() {
    asm volatile("s_wait_asynccnt 0x0" ::: "memory");
}

// ---------------------------------------------------------------------------
// Kernel 0: Wp f32 -> f16
// ---------------------------------------------------------------------------
__global__ void __launch_bounds__(256) convert_f32_to_f16(const float* __restrict__ src,
                                                          f16* __restrict__ dst, int n) {
    int i = blockIdx.x * 256 + threadIdx.x;
    if (i < n) dst[i] = (f16)src[i];
}

// ---------------------------------------------------------------------------
// Kernel 1: QKV projection. One wave per (b*h, 16-row s-tile).
//   Q,K stored [bh][s][d] f16 ; V stored transposed [bh][d][s] f16.
// ---------------------------------------------------------------------------
__global__ void __launch_bounds__(32) qkv_kernel(const float* __restrict__ x,
        const float* __restrict__ Wq, const float* __restrict__ bq,
        const float* __restrict__ Wk, const float* __restrict__ bk,
        const float* __restrict__ Wv, const float* __restrict__ bv,
        f16* __restrict__ Q, f16* __restrict__ Kb, f16* __restrict__ Vt) {
    const int bh = blockIdx.x;          // 0..63
    const int s0 = blockIdx.y * 16;     // row tile base
    const int b = bh >> 4, h = bh & 15;
    const int lane = threadIdx.x & 31;
    const int col = lane & 15, hi = lane >> 4;

    const float* xbase = x + ((size_t)(b * SS + s0)) * EE + h * DD;
    const v16h a0 = load_tile_f32(xbase,      EE);   // d = 0..31
    const v16h a1 = load_tile_f32(xbase + 32, EE);   // d = 32..63

    const float* Ws[3] = {Wq, Wk, Wv};
    const float* bs[3] = {bq, bk, bv};
#pragma unroll
    for (int w = 0; w < 3; ++w) {
#pragma unroll
        for (int n = 0; n < 4; ++n) {
            float bval = bs[w][n * 16 + col];
            v8f c = {};
#pragma unroll
            for (int g = 0; g < 8; ++g) c[g] = bval;
            v16h b0 = load_tile_f32(Ws[w] + (size_t)(n * 16) * DD,      DD);
            v16h b1 = load_tile_f32(Ws[w] + (size_t)(n * 16) * DD + 32, DD);
            c = wmma_f16(a0, b0, c);
            c = wmma_f16(a1, b1, c);
#pragma unroll
            for (int g = 0; g < 8; ++g) {
                int srow = s0 + g + 8 * hi;
                int d = n * 16 + col;
                f16 val = (f16)c[g];
                if (w == 0)      Q [((size_t)bh * SS + srow) * DD + d] = val;
                else if (w == 1) Kb[((size_t)bh * SS + srow) * DD + d] = val;
                else             Vt[((size_t)bh * DD + d) * SS + srow] = val;
            }
        }
    }
}

// ---------------------------------------------------------------------------
// Kernel 2: flash attention. 8 waves / block; 128 query rows share each
// 32-key K/V tile, double-buffered in LDS via async global->LDS loads.
// ---------------------------------------------------------------------------
#define KSTR 72   // LDS stride (halves) for 32x64 K tile: 144B, odd bank step
#define VSTR 40   // LDS stride for 64x32 V tile: 80B
#define PSTR 36   // LDS stride for per-wave 16x32 P staging: 72B
#define KSZ  (32 * KSTR)            // 2304 halves per K buffer
#define VSZ  (64 * VSTR)            // 2560 halves per V buffer
#define POFF (2 * KSZ + 2 * VSZ)    // 9728

__global__ void __launch_bounds__(256) flash_attn_kernel(const f16* __restrict__ Q,
        const f16* __restrict__ Kb, const f16* __restrict__ Vt,
        f16* __restrict__ attn) {
    const int bh = blockIdx.x;
    const int b = bh >> 4, h = bh & 15;
    const int tid  = threadIdx.x;
    const int w    = tid >> 5;
    const int lane = tid & 31;
    const int col = lane & 15, hi = lane >> 4;
    const int s0 = blockIdx.y * 128 + w * 16;       // this wave's query rows

    __shared__ __align__(16) f16 smem[POFF + 8 * 16 * PSTR];   // ~28 KB
    f16* ldsP = smem + POFF + w * (16 * PSTR);

    // Per-thread async-copy assignment: 256 threads x 16B covers each 4KB tile.
    const int krow = tid >> 3, kchunk = tid & 7;    // K: 32 rows x 128B
    const int vrow = tid >> 2, vchunk = tid & 3;    // V: 64 rows x 64B

    auto issue_tiles = [&](int j, int buf) {
        f16* kbuf = smem + buf * KSZ;               // no pointer arrays: compute
        f16* vbuf = smem + 2 * KSZ + buf * VSZ;     // bases arithmetically
        async_b128(Kb + ((size_t)bh * SS + j * 32 + krow) * DD + kchunk * 8,
                   kbuf + krow * KSTR + kchunk * 8);
        async_b128(Vt + ((size_t)bh * DD + vrow) * SS + j * 32 + vchunk * 8,
                   vbuf + vrow * VSTR + vchunk * 8);
    };

    const f16* qbase = Q + ((size_t)bh * SS + s0) * DD;
    const v16h aq0 = load_tile_f16(qbase,      DD);
    const v16h aq1 = load_tile_f16(qbase + 32, DD);

    float mrow[8], lrow[8];
    v8f oacc[4] = {};
#pragma unroll
    for (int g = 0; g < 8; ++g) { mrow[g] = -3.0e38f; lrow[g] = 0.0f; }

    issue_tiles(0, 0);
    for (int j = 0; j < SS / 32; ++j) {
        const int buf = j & 1;
        const f16* kcur = smem + buf * KSZ;
        const f16* vcur = smem + 2 * KSZ + buf * VSZ;

        wait_async0();           // my async copies for tile j done
        __syncthreads();         // everyone's copies done; prior reads retired
        if (j + 1 < SS / 32) issue_tiles(j + 1, buf ^ 1);   // overlap next tile

        // ---- scores: Q(16x64) x K^T(64x32) as two 16x16 tiles ----
        float p[2][8];
#pragma unroll
        for (int t = 0; t < 2; ++t) {
            const f16* kbase = kcur + t * 16 * KSTR;
            v16h b0 = load_tile_f16(kbase,      KSTR);
            v16h b1 = load_tile_f16(kbase + 32, KSTR);
            v8f c = {};
            c = wmma_f16(aq0, b0, c);
            c = wmma_f16(aq1, b1, c);
#pragma unroll
            for (int g = 0; g < 8; ++g) p[t][g] = c[g] * 0.125f; // 1/sqrt(64)
        }
        // ---- online softmax (rows live in 16-lane halves) ----
#pragma unroll
        for (int g = 0; g < 8; ++g) {
            float tmax = fmaxf(p[0][g], p[1][g]);
#pragma unroll
            for (int msk = 1; msk < 16; msk <<= 1)
                tmax = fmaxf(tmax, __shfl_xor(tmax, msk, 16));
            float mnew  = fmaxf(mrow[g], tmax);
            float scale = __expf(mrow[g] - mnew);
            mrow[g] = mnew;
            p[0][g] = __expf(p[0][g] - mnew);
            p[1][g] = __expf(p[1][g] - mnew);
            float ps = p[0][g] + p[1][g];
#pragma unroll
            for (int msk = 1; msk < 16; msk <<= 1)
                ps += __shfl_xor(ps, msk, 16);
            lrow[g] = lrow[g] * scale + ps;
#pragma unroll
            for (int n = 0; n < 4; ++n) oacc[n][g] *= scale;
        }
        // ---- re-stripe P into A-operand layout via per-wave LDS region ----
        // (DS ops execute in order within a wave; only compiler ordering needed)
#pragma unroll
        for (int g = 0; g < 8; ++g) {
            int m = g + 8 * hi;
            ldsP[m * PSTR + col]      = (f16)p[0][g];
            ldsP[m * PSTR + 16 + col] = (f16)p[1][g];
        }
        asm volatile("" ::: "memory");
        v16h ap = load_tile_f16(ldsP, PSTR);
        // ---- P(16x32) x V(32x16) for each of 4 d-tiles ----
#pragma unroll
        for (int n = 0; n < 4; ++n) {
            v16h bv = load_tile_f16(vcur + (n * 16) * VSTR, VSTR);
            oacc[n] = wmma_f16(ap, bv, oacc[n]);
        }
    }
    // normalize and store attn as [b][s][h*64+d] f16
#pragma unroll
    for (int g = 0; g < 8; ++g) {
        float rl = 1.0f / lrow[g];
        int srow = s0 + g + 8 * hi;
#pragma unroll
        for (int n = 0; n < 4; ++n) {
            attn[((size_t)b * SS + srow) * EE + h * DD + n * 16 + col] =
                (f16)(oacc[n][g] * rl);
        }
    }
}

// ---------------------------------------------------------------------------
// Kernel 3: output projection  out = attn @ Wp^T + bp   (f32 out)
// 8 waves / block compute a 128x128 tile; A and B K-chunks staged in LDS.
// ---------------------------------------------------------------------------
#define OSTR 40   // LDS stride (halves) for 128x32 chunks: 80B, conflict-free

__global__ void __launch_bounds__(256) out_proj_kernel(const f16* __restrict__ attn,
        const f16* __restrict__ Wp16, const float* __restrict__ bp,
        float* __restrict__ out) {
    const int rt = blockIdx.x;                      // 0..63 (128-row groups)
    const int ct = blockIdx.y;                      // 0..7  (128-col groups)
    const int tid  = threadIdx.x;
    const int w    = tid >> 5;
    const int lane = tid & 31;
    const int col = lane & 15, hi = lane >> 4;

    __shared__ __align__(16) f16 smem2[2 * 128 * OSTR];        // 20 KB
    f16* ldsA = smem2;
    f16* ldsB = smem2 + 128 * OSTR;

    v8f acc[8];
#pragma unroll
    for (int c2 = 0; c2 < 8; ++c2) {
        float bval = bp[ct * 128 + c2 * 16 + col];
#pragma unroll
        for (int g = 0; g < 8; ++g) acc[c2][g] = bval;
    }

    const int lrow = tid >> 2;       // 0..63 (two passes cover 128 rows)
    const int lchunk = tid & 3;      // 16B chunks of the 64B-wide k-slab
    for (int kc = 0; kc < EE / 32; ++kc) {
        __syncthreads();             // previous chunk fully consumed
#pragma unroll
        for (int rr = 0; rr < 2; ++rr) {
            int row = lrow + 64 * rr;
            const int4* asrc = (const int4*)(attn +
                ((size_t)(rt * 128 + row)) * EE + kc * 32 + lchunk * 8);
            *(int4*)(ldsA + row * OSTR + lchunk * 8) = *asrc;
            const int4* bsrc = (const int4*)(Wp16 +
                ((size_t)(ct * 128 + row)) * EE + kc * 32 + lchunk * 8);
            *(int4*)(ldsB + row * OSTR + lchunk * 8) = *bsrc;
        }
        __syncthreads();
        v16h a = load_tile_f16(ldsA + w * 16 * OSTR, OSTR);
#pragma unroll
        for (int c2 = 0; c2 < 8; ++c2) {
            v16h bm = load_tile_f16(ldsB + c2 * 16 * OSTR, OSTR);
            acc[c2] = wmma_f16(a, bm, acc[c2]);
        }
    }
#pragma unroll
    for (int c2 = 0; c2 < 8; ++c2) {
#pragma unroll
        for (int g = 0; g < 8; ++g) {
            int row = rt * 128 + w * 16 + g + 8 * hi;
            out[(size_t)row * EE + ct * 128 + c2 * 16 + col] = acc[c2][g];
        }
    }
}

// ---------------------------------------------------------------------------
extern "C" void kernel_launch(void* const* d_in, const int* in_sizes, int n_in,
                              void* d_out, int out_size, void* d_ws, size_t ws_size,
                              hipStream_t stream) {
    const float* x  = (const float*)d_in[0];
    const float* Wq = (const float*)d_in[1];
    const float* bq = (const float*)d_in[2];
    const float* Wk = (const float*)d_in[3];
    const float* bk = (const float*)d_in[4];
    const float* Wv = (const float*)d_in[5];
    const float* bv = (const float*)d_in[6];
    const float* Wp = (const float*)d_in[7];
    const float* bp = (const float*)d_in[8];
    float* out = (float*)d_out;

    const size_t nPH = (size_t)BB * HH * SS * DD;   // 8,388,608 halves / tensor
    f16* Q    = (f16*)d_ws;
    f16* Kb   = Q  + nPH;
    f16* Vt   = Kb + nPH;
    f16* attn = Vt + nPH;
    f16* Wp16 = attn + (size_t)BB * SS * EE;        // ~66 MB total workspace

    convert_f32_to_f16<<<(EE * EE + 255) / 256, 256, 0, stream>>>(Wp, Wp16, EE * EE);
    qkv_kernel<<<dim3(BB * HH, SS / 16), 32, 0, stream>>>(x, Wq, bq, Wk, bk, Wv, bv,
                                                          Q, Kb, Vt);
    flash_attn_kernel<<<dim3(BB * HH, SS / 128), 256, 0, stream>>>(Q, Kb, Vt, attn);
    out_proj_kernel<<<dim3(BB * SS / 128, EE / 128), 256, 0, stream>>>(attn, Wp16, bp, out);
}
